// SolidMechanicsGNN_V3_14096082665484
// MI455X (gfx1250) — matrix operand order repaired and checked by reading, not compiled, not run on hardware
//
#include <hip/hip_runtime.h>
#include <hip/hip_bf16.h>

#define GNN_N   10000
#define GNN_E   320000
#define GNN_D   128
#define GNN_L   3
#define GNN_IN  5
#define LN_EPS  1e-5f

typedef _Float16 half_t;
typedef half_t v8h  __attribute__((ext_vector_type(8)));
typedef half_t v16h __attribute__((ext_vector_type(16)));
typedef float  v8f  __attribute__((ext_vector_type(8)));

// --------------------------------------------------------------------------
// Optional CDNA5 async global->LDS path (ASYNCcnt tracked), with fallback.
// Builtin signature (probed from diagnostics): takes pointers to
// "int __attribute__((vector_size(16)))" in AS(1) / AS(3), plus imm offset+cpol.
// --------------------------------------------------------------------------
#if defined(__has_builtin)
#if __has_builtin(__builtin_amdgcn_global_load_async_to_lds_b128) && \
    __has_builtin(__builtin_amdgcn_s_wait_asynccnt)
#define HAVE_ASYNC_LDS 1
#endif
#endif

typedef int v4i_vs __attribute__((vector_size(16)));
typedef __attribute__((address_space(1))) v4i_vs glb_v4i;
typedef __attribute__((address_space(3))) v4i_vs lds_v4i;

__device__ __forceinline__ void copy16_g2l(const half_t* __restrict__ g,
                                           half_t* l) {
#ifdef HAVE_ASYNC_LDS
  __builtin_amdgcn_global_load_async_to_lds_b128((glb_v4i*)g, (lds_v4i*)l, 0, 0);
#else
  *(v8h*)l = *(const v8h*)g;
#endif
}

__device__ __forceinline__ void wait_async_copies() {
#ifdef HAVE_ASYNC_LDS
  __builtin_amdgcn_s_wait_asynccnt(0);
#endif
}

// ---------------------------------------------------------------------------
// WMMA helpers (CDNA5 V_WMMA_F32_16X16X32_F16, wave32)
// ---------------------------------------------------------------------------
__device__ __forceinline__ v8f wmma_f16(v16h a, v16h b, v8f c) {
  return __builtin_amdgcn_wmma_f32_16x16x32_f16(false, a, false, b, (short)0, c,
                                                false, false);
}

// A-fragment from an LDS tile (row-major, stride in halves).
// 16-bit A 16x32 layout: lanes<16: K=k0+0..7 & k0+16..23,
// lanes>=16: K=k0+8..15 & k0+24..31 -> two contiguous ds_load_b128.
__device__ __forceinline__ v16h lds_a_frag(const half_t* lds, int stride_h,
                                           int m, int k0, int khalf) {
  const half_t* p = lds + m * stride_h + k0 + khalf * 8;
  union { v16h v; v8h h[2]; } u;
  u.h[0] = *(const v8h*)(p);
  u.h[1] = *(const v8h*)(p + 16);
  return u.v;
}

// B-fragment from row-major f32 W[k][n] (ld=128), converted to f16.
// 16-bit B 32x16 layout: lanes<16 K=k0+0..7 & k0+16..23, lanes>=16 K=k0+8..15 & k0+24..31.
__device__ __forceinline__ v16h glb_b_frag(const float* __restrict__ W, int ld,
                                           int k0, int kmax, int ncol, int klane) {
  v16h b;
#pragma unroll
  for (int j = 0; j < 16; ++j) {
    int k = k0 + (klane >> 1) + (j & 7) + ((j >> 3) << 4);
    b[j] = (k < kmax) ? (half_t)W[(size_t)k * ld + ncol] : (half_t)0.0f;
  }
  return b;
}

// ---------------------------------------------------------------------------
// Encoder: h = relu(x @ enc_w + enc_b); also emit f16 copy of h.
// ---------------------------------------------------------------------------
__global__ void encoder_kernel(const float* __restrict__ x,
                               const float* __restrict__ w,
                               const float* __restrict__ b,
                               float* __restrict__ h,
                               half_t* __restrict__ hh) {
  int i = blockIdx.x * blockDim.x + threadIdx.x;
  if (i >= GNN_N * GNN_D) return;
  int node = i >> 7, d = i & 127;
  float acc = b[d];
#pragma unroll
  for (int k = 0; k < GNN_IN; ++k)
    acc += x[node * GNN_IN + k] * w[k * GNN_D + d];
  acc = fmaxf(acc, 0.0f);
  h[i] = acc;
  hh[i] = (half_t)acc;
}

// ---------------------------------------------------------------------------
// Edge message kernel, 32 edges per tile (2 x 16-row WMMA sub-tiles).
//   m = relu(LN(concat(h[row], h[col], eattr) @ W1 + b1))
//   m = relu(m @ W2 + b2);  aggr[row] += m (atomic scatter-add)
// 8 waves; wave w owns output columns [16w, 16w+16). Each wave runs two
// independent load->wmma chains (sub-tiles), so LDS latency overlaps WMMA.
// ---------------------------------------------------------------------------
__launch_bounds__(256, 1)
__global__ void edge_msg_kernel(const half_t* __restrict__ hh,
                                const int* __restrict__ erow,
                                const int* __restrict__ ecol,
                                const float* __restrict__ eattr,
                                const float* __restrict__ w1,
                                const float* __restrict__ b1,
                                const float* __restrict__ g1,
                                const float* __restrict__ be1,
                                const float* __restrict__ w2,
                                const float* __restrict__ b2,
                                float* __restrict__ aggr) {
  __shared__ __align__(16) half_t Atile[32 * 288];  // 32 edges x (258 pad 288)
  __shared__ __align__(16) half_t M1[32 * 128];
  __shared__ float rowsum[32];
  __shared__ float rowsumsq[32];
  __shared__ int   srow[32];

  const int tid   = threadIdx.x;
  const int lane  = tid & 31;
  const int wave  = tid >> 5;
  const int khalf = lane >> 4;
  const int klane = khalf * 16;
  const int mrow  = lane & 15;
  const int ncol  = (lane & 15) + wave * 16;

  // Persistent per-wave weight fragments (reused across all edge tiles).
  v16h B1[9];
#pragma unroll
  for (int c = 0; c < 9; ++c) B1[c] = glb_b_frag(w1, GNN_D, c * 32, 258, ncol, klane);
  v16h B2[4];
#pragma unroll
  for (int c = 0; c < 4; ++c) B2[c] = glb_b_frag(w2, GNN_D, c * 32, 128, ncol, klane);
  const float bias1 = b1[ncol], gam = g1[ncol], bet = be1[ncol], bias2 = b2[ncol];

  const int ntiles = GNN_E / 32;
  for (int tile = blockIdx.x; tile < ntiles; tile += gridDim.x) {
    const int e0 = tile * 32;

    // ---- stage A tile: [h_row | h_col | eattr | zero-pad] in f16 ----
#pragma unroll
    for (int rep = 0; rep < 2; ++rep) {
      const int idx = tid + rep * 256;
      const int e = idx >> 4;          // 0..31
      const int p = idx & 15;          // 16B chunk within the 128-half row
      const int r = erow[e0 + e];
      const int c = ecol[e0 + e];
      if (p == 0) srow[e] = r;
      copy16_g2l(hh + (size_t)r * GNN_D + p * 8, Atile + e * 288 + p * 8);
      copy16_g2l(hh + (size_t)c * GNN_D + p * 8, Atile + e * 288 + 128 + p * 8);
    }
    {
      const int e = tid & 31;
      if (tid < 32) {
        Atile[e * 288 + 256] = (half_t)eattr[(size_t)(e0 + e) * 2 + 0];
        Atile[e * 288 + 257] = (half_t)eattr[(size_t)(e0 + e) * 2 + 1];
#pragma unroll
        for (int z = 258; z < 288; ++z) Atile[e * 288 + z] = (half_t)0.0f;
        rowsum[e] = 0.0f;
        rowsumsq[e] = 0.0f;
      }
      if (tile + gridDim.x < ntiles && tid >= 32 && tid < 64)
        __builtin_prefetch(&erow[e0 + (int)gridDim.x * 32 + (tid - 32)], 0, 3);
    }
    wait_async_copies();
    __syncthreads();

    // ---- GEMM1: [32 x 288] @ [288 x 128], two independent sub-tiles ----
    v8f acc0 = {0.f, 0.f, 0.f, 0.f, 0.f, 0.f, 0.f, 0.f};
    v8f acc1 = {0.f, 0.f, 0.f, 0.f, 0.f, 0.f, 0.f, 0.f};
#pragma unroll
    for (int c = 0; c < 9; ++c) {
      v16h a0 = lds_a_frag(Atile, 288, mrow,      c * 32, khalf);
      v16h a1 = lds_a_frag(Atile, 288, mrow + 16, c * 32, khalf);
      acc0 = wmma_f16(a0, B1[c], acc0);
      acc1 = wmma_f16(a1, B1[c], acc1);
    }

    // bias, then LN row statistics via LDS atomics
#pragma unroll
    for (int r = 0; r < 8; ++r) {
      float v0 = acc0[r] + bias1;
      float v1 = acc1[r] + bias1;
      acc0[r] = v0;
      acc1[r] = v1;
      int M = r + 8 * khalf;
      atomicAdd(&rowsum[M], v0);
      atomicAdd(&rowsumsq[M], v0 * v0);
      atomicAdd(&rowsum[M + 16], v1);
      atomicAdd(&rowsumsq[M + 16], v1 * v1);
    }
    __syncthreads();

    // ---- normalize + affine + relu -> M1 (f16 in LDS) ----
#pragma unroll
    for (int r = 0; r < 8; ++r) {
#pragma unroll
      for (int s = 0; s < 2; ++s) {
        int M = r + 8 * khalf + 16 * s;
        float v = (s == 0) ? acc0[r] : acc1[r];
        float mean = rowsum[M] * (1.0f / 128.0f);
        float var  = rowsumsq[M] * (1.0f / 128.0f) - mean * mean;
        float rstd = rsqrtf(var + LN_EPS);
        float y = fmaxf((v - mean) * rstd * gam + bet, 0.0f);
        M1[M * 128 + ncol] = (half_t)y;
      }
    }
    __syncthreads();

    // ---- GEMM2: [32 x 128] @ [128 x 128] ----
    v8f d0 = {0.f, 0.f, 0.f, 0.f, 0.f, 0.f, 0.f, 0.f};
    v8f d1 = {0.f, 0.f, 0.f, 0.f, 0.f, 0.f, 0.f, 0.f};
#pragma unroll
    for (int c = 0; c < 4; ++c) {
      v16h a0 = lds_a_frag(M1, 128, mrow,      c * 32, khalf);
      v16h a1 = lds_a_frag(M1, 128, mrow + 16, c * 32, khalf);
      d0 = wmma_f16(a0, B2[c], d0);
      d1 = wmma_f16(a1, B2[c], d1);
    }

    // ---- relu + scatter-add onto source nodes ----
#pragma unroll
    for (int r = 0; r < 8; ++r) {
      int M = r + 8 * khalf;
      float y0 = fmaxf(d0[r] + bias2, 0.0f);
      float y1 = fmaxf(d1[r] + bias2, 0.0f);
      atomicAdd(&aggr[(size_t)srow[M] * GNN_D + ncol], y0);
      atomicAdd(&aggr[(size_t)srow[M + 16] * GNN_D + ncol], y1);
    }
    __syncthreads();  // protect Atile/srow/rowsum reuse in next iteration
  }
}

// ---------------------------------------------------------------------------
// Node update kernel: u = relu(LN(concat(h, aggr) @ Wup + bup)); h += u
// ---------------------------------------------------------------------------
__launch_bounds__(256, 1)
__global__ void node_update_kernel(const float* __restrict__ wu,
                                   const float* __restrict__ bu,
                                   const float* __restrict__ g,
                                   const float* __restrict__ be,
                                   const float* __restrict__ aggr,
                                   float* __restrict__ h,
                                   half_t* __restrict__ hh) {
  __shared__ __align__(16) half_t Atile[16 * 256];
  __shared__ float rowsum[16];
  __shared__ float rowsumsq[16];

  const int tid   = threadIdx.x;
  const int lane  = tid & 31;
  const int wave  = tid >> 5;
  const int khalf = lane >> 4;
  const int klane = khalf * 16;
  const int mrow  = lane & 15;
  const int ncol  = (lane & 15) + wave * 16;
  const int node0 = blockIdx.x * 16;

  v16h BU[8];
#pragma unroll
  for (int c = 0; c < 8; ++c) BU[c] = glb_b_frag(wu, GNN_D, c * 32, 256, ncol, klane);
  const float bias = bu[ncol], gam = g[ncol], bet = be[ncol];

  // ---- stage A tile: [h (async f16 copy) | aggr (f32->f16 convert)] ----
  {
    const int nd = tid >> 4;
    const int p  = tid & 15;
    copy16_g2l(hh + (size_t)(node0 + nd) * GNN_D + p * 8, Atile + nd * 256 + p * 8);
    const float4* src = (const float4*)(aggr + (size_t)(node0 + nd) * GNN_D) + p * 2;
    float4 a0 = src[0], a1 = src[1];
    half_t* dst = Atile + nd * 256 + 128 + p * 8;
    dst[0] = (half_t)a0.x; dst[1] = (half_t)a0.y; dst[2] = (half_t)a0.z; dst[3] = (half_t)a0.w;
    dst[4] = (half_t)a1.x; dst[5] = (half_t)a1.y; dst[6] = (half_t)a1.z; dst[7] = (half_t)a1.w;
    if (tid < 16) { rowsum[tid] = 0.0f; rowsumsq[tid] = 0.0f; }
  }
  wait_async_copies();
  __syncthreads();

  // ---- GEMM: [16 x 256] @ [256 x 128] ----
  v8f acc = {0.f, 0.f, 0.f, 0.f, 0.f, 0.f, 0.f, 0.f};
#pragma unroll
  for (int c = 0; c < 8; ++c)
    acc = wmma_f16(lds_a_frag(Atile, 256, mrow, c * 32, khalf), BU[c], acc);

#pragma unroll
  for (int r = 0; r < 8; ++r) {
    float v = acc[r] + bias;
    acc[r] = v;
    int M = r + 8 * khalf;
    atomicAdd(&rowsum[M], v);
    atomicAdd(&rowsumsq[M], v * v);
  }
  __syncthreads();

  // ---- LN + relu + residual, write back f32 and f16 h ----
#pragma unroll
  for (int r = 0; r < 8; ++r) {
    int M = r + 8 * khalf;
    float mean = rowsum[M] * (1.0f / 128.0f);
    float var  = rowsumsq[M] * (1.0f / 128.0f) - mean * mean;
    float rstd = rsqrtf(var + LN_EPS);
    float u = fmaxf((acc[r] - mean) * rstd * gam + bet, 0.0f);
    size_t idx = (size_t)(node0 + M) * GNN_D + ncol;
    float hn = h[idx] + u;
    h[idx]  = hn;
    hh[idx] = (half_t)hn;
  }
}

// ---------------------------------------------------------------------------
// Decoder: mid = relu(h @ dec1 + b1); out = mid @ dec2 + b2
// ---------------------------------------------------------------------------
__global__ void dec1_kernel(const float* __restrict__ h, const float* __restrict__ w,
                            const float* __restrict__ b, float* __restrict__ mid) {
  int i = blockIdx.x * blockDim.x + threadIdx.x;
  if (i >= GNN_N * 64) return;
  int node = i >> 6, j = i & 63;
  float acc = b[j];
  const float* hr = h + (size_t)node * GNN_D;
#pragma unroll 8
  for (int k = 0; k < GNN_D; ++k) acc += hr[k] * w[k * 64 + j];
  mid[i] = fmaxf(acc, 0.0f);
}

__global__ void dec2_kernel(const float* __restrict__ mid, const float* __restrict__ w,
                            const float* __restrict__ b, float* __restrict__ out) {
  int i = blockIdx.x * blockDim.x + threadIdx.x;
  if (i >= GNN_N * 2) return;
  int node = i >> 1, j = i & 1;
  float acc = b[j];
  const float* mr = mid + (size_t)node * 64;
#pragma unroll 8
  for (int k = 0; k < 64; ++k) acc += mr[k] * w[k * 2 + j];
  out[i] = acc;
}

// ---------------------------------------------------------------------------
// Host launcher
// ---------------------------------------------------------------------------
extern "C" void kernel_launch(void* const* d_in, const int* in_sizes, int n_in,
                              void* d_out, int out_size, void* d_ws, size_t ws_size,
                              hipStream_t stream) {
  const float* x_batch = (const float*)d_in[0];
  const int*   eidx    = (const int*)d_in[1];
  const float* eattr   = (const float*)d_in[2];
  const float* enc_w   = (const float*)d_in[3];
  const float* enc_b   = (const float*)d_in[4];
  const float* msg1_w  = (const float*)d_in[5];
  const float* msg1_b  = (const float*)d_in[6];
  const float* mln_g   = (const float*)d_in[7];
  const float* mln_b   = (const float*)d_in[8];
  const float* msg2_w  = (const float*)d_in[9];
  const float* msg2_b  = (const float*)d_in[10];
  const float* up_w    = (const float*)d_in[11];
  const float* up_b    = (const float*)d_in[12];
  const float* uln_g   = (const float*)d_in[13];
  const float* uln_b   = (const float*)d_in[14];
  const float* dec1_w  = (const float*)d_in[15];
  const float* dec1_b  = (const float*)d_in[16];
  const float* dec2_w  = (const float*)d_in[17];
  const float* dec2_b  = (const float*)d_in[18];

  const int* erow = eidx;
  const int* ecol = eidx + GNN_E;

  const size_t nd = (size_t)GNN_N * GNN_D;
  float*  h    = (float*)d_ws;
  half_t* hh   = (half_t*)((char*)d_ws + 2 * nd * sizeof(float));
  float*  aggr = (float*)((char*)hh + 2 * nd * sizeof(half_t));
  float*  mid  = (float*)((char*)aggr + 2 * nd * sizeof(float));
  float*  out  = (float*)d_out;

  for (int b = 0; b < 2; ++b) {
    float*  hb  = h    + (size_t)b * nd;
    half_t* hhb = hh   + (size_t)b * nd;
    float*  ab  = aggr + (size_t)b * nd;
    float*  mb  = mid  + (size_t)b * GNN_N * 64;

    encoder_kernel<<<(GNN_N * GNN_D + 255) / 256, 256, 0, stream>>>(
        x_batch + (size_t)b * GNN_N * GNN_IN, enc_w, enc_b, hb, hhb);

    for (int l = 0; l < GNN_L; ++l) {
      (void)hipMemsetAsync(ab, 0, nd * sizeof(float), stream);
      edge_msg_kernel<<<640, 256, 0, stream>>>(
          hhb, erow, ecol, eattr,
          msg1_w + (size_t)l * 258 * GNN_D, msg1_b + (size_t)l * GNN_D,
          mln_g  + (size_t)l * GNN_D,       mln_b  + (size_t)l * GNN_D,
          msg2_w + (size_t)l * GNN_D * GNN_D, msg2_b + (size_t)l * GNN_D,
          ab);
      node_update_kernel<<<GNN_N / 16, 256, 0, stream>>>(
          up_w + (size_t)l * 256 * GNN_D, up_b + (size_t)l * GNN_D,
          uln_g + (size_t)l * GNN_D, uln_b + (size_t)l * GNN_D,
          ab, hb, hhb);
    }

    dec1_kernel<<<(GNN_N * 64 + 255) / 256, 256, 0, stream>>>(hb, dec1_w, dec1_b, mb);
    dec2_kernel<<<(GNN_N * 2 + 255) / 256, 256, 0, stream>>>(mb, dec2_w, dec2_b,
                                                             out + (size_t)b * GNN_N * 2);
  }
}
